// MoETransformerBlock_43894565765188
// MI455X (gfx1250) — matrix operand order repaired and checked by reading, not compiled, not run on hardware
//
#include <hip/hip_runtime.h>
#include <hip/hip_bf16.h>

// ---------------- types & helpers ----------------
typedef __attribute__((ext_vector_type(16))) __bf16 v16bf;
typedef __attribute__((ext_vector_type(8)))  float  v8f;
typedef __attribute__((ext_vector_type(4)))  unsigned int u32x4;

struct FragBF {
  union { v16bf v; u32x4 q[2]; unsigned short u[16]; };
};

__device__ __forceinline__ unsigned short f2bf(float x) {
  unsigned int b = __float_as_uint(x);
  unsigned int r = b + 0x7FFFu + ((b >> 16) & 1u);
  return (unsigned short)(r >> 16);
}
__device__ __forceinline__ float bf2f(unsigned short h) {
  return __uint_as_float(((unsigned int)h) << 16);
}

// A fragment: 16x32 bf16, row-major source, row = lane&15 (caller passes row ptr)
__device__ __forceinline__ void load_a_frag(FragBF& f, const unsigned short* rowPtr, int k0, int lane) {
  const int kk = k0 + ((lane & 16) ? 8 : 0);
  f.q[0] = *reinterpret_cast<const u32x4*>(rowPtr + kk);
  f.q[1] = *reinterpret_cast<const u32x4*>(rowPtr + kk + 16);
}
// B fragment: 32x16 bf16, row-major source (ldn in elements); lane%16 = N col, elements = K (table order)
__device__ __forceinline__ void load_b_frag_bf16(FragBF& f, const unsigned short* B, int ldn, int k0, int n0, int lane) {
  const unsigned short* p = B + (size_t)(k0 + ((lane & 16) ? 8 : 0)) * ldn + n0 + (lane & 15);
#pragma unroll
  for (int s = 0; s < 8; ++s) {
    f.u[s] = p[(size_t)s * ldn];
    f.u[8 + s] = p[(size_t)(16 + s) * ldn];
  }
}
// B fragment from f32 source with on-the-fly bf16 conversion (fallback path)
__device__ __forceinline__ void load_b_frag_f32(FragBF& f, const float* B, int ldn, int k0, int n0, int lane) {
  const float* p = B + (size_t)(k0 + ((lane & 16) ? 8 : 0)) * ldn + n0 + (lane & 15);
#pragma unroll
  for (int s = 0; s < 8; ++s) {
    f.u[s] = f2bf(p[(size_t)s * ldn]);
    f.u[8 + s] = f2bf(p[(size_t)(16 + s) * ldn]);
  }
}
template <bool PRE>
__device__ __forceinline__ void load_b(FragBF& f, const float* Bf, const unsigned short* Bh,
                                       int ldn, int k0, int n0, int lane) {
  if constexpr (PRE) load_b_frag_bf16(f, Bh, ldn, k0, n0, lane);
  else load_b_frag_f32(f, Bf, ldn, k0, n0, lane);
}
__device__ __forceinline__ v8f wmma_bf16(const FragBF& a, const FragBF& b, v8f c) {
  return __builtin_amdgcn_wmma_f32_16x16x32_bf16(false, a.v, false, b.v, (short)0, c, false, false);
}

// --- CDNA5 async global->LDS staging of a 32(K) x 64(N) bf16 tile (4 KB) ---
// 128 threads: thread t copies 32B of row (t>>2), columns (t&3)*16..+15 (two b128 async ops)
__device__ __forceinline__ void stage_b_async(unsigned short* lbuf, const unsigned short* Bh,
                                              int ldn, int k0, int n0, int tid) {
  const int row = tid >> 2;
  const int ce = (tid & 3) * 16;
  const unsigned short* g = Bh + (size_t)(k0 + row) * ldn + n0 + ce;
  const unsigned int lo = (unsigned int)(uintptr_t)(lbuf + row * 64 + ce);
  asm volatile("global_load_async_to_lds_b128 %0, %1, off" :: "v"(lo), "v"(g) : "memory");
  asm volatile("global_load_async_to_lds_b128 %0, %1, off" :: "v"(lo + 16), "v"(g + 8) : "memory");
}
__device__ __forceinline__ void wait_async0() {
  asm volatile("s_wait_asynccnt 0" ::: "memory");
}
// B fragment from staged LDS tile (ld = 64 elements): lane%16 = N col, elements = K (table order)
__device__ __forceinline__ void load_b_frag_lds(FragBF& f, const unsigned short* lbuf, int j, int lane) {
  const unsigned short* p = lbuf + ((lane & 16) ? 8 : 0) * 64 + 16 * j + (lane & 15);
#pragma unroll
  for (int s = 0; s < 8; ++s) {
    f.u[s] = p[s * 64];
    f.u[8 + s] = p[(16 + s) * 64];
  }
}

// ---------------- model constants ----------------
#define HID 768
#define NQ 12
#define NKV 3
#define HD 64
#define SEQ 2048
#define QKVN 1152   // (NQ+2*NKV)*HD
#define NEXP 16
#define FF 1536
#define MAXTILES 288   // >= 4096/16 + 16
#define PADSLOTS 4352  // 4096 + 16*16

// sched int-array layout
#define SCH_COUNT 0
#define SCH_PADOFF 16
#define SCH_CURSOR 32
#define SCH_TOTAL 48
#define SCH_TEXP 64
#define SCH_TSLOT (64 + MAXTILES)
#define SCH_INTS (64 + 2 * MAXTILES)

// ---------------- small kernels ----------------

__global__ void k_cvt_bf16(const float* __restrict__ src, unsigned short* __restrict__ dst, int n4) {
  const int stride = gridDim.x * blockDim.x;
  for (int i = blockIdx.x * blockDim.x + threadIdx.x; i < n4; i += stride) {
    float4 t = reinterpret_cast<const float4*>(src)[i];
    uint2 o;
    o.x = (unsigned)f2bf(t.x) | ((unsigned)f2bf(t.y) << 16);
    o.y = (unsigned)f2bf(t.z) | ((unsigned)f2bf(t.w) << 16);
    reinterpret_cast<uint2*>(dst)[i] = o;
  }
}

__global__ void k_rmsnorm(const float* __restrict__ x, const float* __restrict__ w,
                          unsigned short* __restrict__ o) {
  __shared__ float red[256];
  const int t = blockIdx.x;
  const float* xr = x + (size_t)t * HID;
  float ss = 0.f;
  for (int i = threadIdx.x; i < HID; i += 256) { float v = xr[i]; ss += v * v; }
  red[threadIdx.x] = ss;
  __syncthreads();
  for (int s = 128; s > 0; s >>= 1) {
    if ((int)threadIdx.x < s) red[threadIdx.x] += red[threadIdx.x + s];
    __syncthreads();
  }
  const float inv = rsqrtf(red[0] / (float)HID + 1e-6f);
  unsigned short* orow = o + (size_t)t * HID;
  for (int i = threadIdx.x; i < HID; i += 256) orow[i] = f2bf(xr[i] * inv * w[i]);
}

// RoPE + precision packing: q -> qb[h][t][d], k -> kT[kv][d][t], v -> vb[kv][t][d]
__global__ void k_rope(const float* __restrict__ qkv, unsigned short* __restrict__ qb,
                       unsigned short* __restrict__ kT, unsigned short* __restrict__ vb) {
  const int g = blockIdx.x * blockDim.x + threadIdx.x;
  if (g >= SEQ * 18) return;
  const int t = g / 18, u = g % 18;
  const float* row = qkv + (size_t)t * QKVN;
  if (u < NQ) {
    const float* src = row + u * HD;
    unsigned short* dst = qb + ((size_t)u * SEQ + t) * HD;
#pragma unroll
    for (int p = 0; p < HD / 2; ++p) {
      float inv = __powf(10000.f, -(float)(2 * p) / (float)HD);
      float sn, cs;
      __sincosf((float)t * inv, &sn, &cs);
      float x0 = src[2 * p], x1 = src[2 * p + 1];
      dst[2 * p] = f2bf(x0 * cs - x1 * sn);
      dst[2 * p + 1] = f2bf(x0 * sn + x1 * cs);
    }
  } else if (u < NQ + NKV) {
    const int kv = u - NQ;
    const float* src = row + NQ * HD + kv * HD;
    unsigned short* base = kT + (size_t)kv * HD * SEQ;
#pragma unroll
    for (int p = 0; p < HD / 2; ++p) {
      float inv = __powf(10000.f, -(float)(2 * p) / (float)HD);
      float sn, cs;
      __sincosf((float)t * inv, &sn, &cs);
      float x0 = src[2 * p], x1 = src[2 * p + 1];
      base[(size_t)(2 * p) * SEQ + t] = f2bf(x0 * cs - x1 * sn);
      base[(size_t)(2 * p + 1) * SEQ + t] = f2bf(x0 * sn + x1 * cs);
    }
  } else {
    const int kv = u - (NQ + NKV);
    const float* src = row + (NQ + NKV) * HD + kv * HD;
    unsigned short* dst = vb + ((size_t)kv * SEQ + t) * HD;
#pragma unroll
    for (int d = 0; d < HD; ++d) dst[d] = f2bf(src[d]);
  }
}

// ---------------- dense GEMMs: block = 64(M) x 64(N), 4 waves share async-staged B ----------------

// qkv = xn1(bf16)[2048x768] @ w_qkv[768x1152] -> f32
template <bool PRE>
__global__ void __launch_bounds__(128) k_gemm_qkv(const unsigned short* __restrict__ A,
                                                  const float* __restrict__ Bf,
                                                  const unsigned short* __restrict__ Bh,
                                                  float* __restrict__ C) {
  __shared__ alignas(16) unsigned short lb[2][32 * 64];
  const int tid = threadIdx.x;
  const int lane = tid & 31;
  const int wv = tid >> 5;
  const int mb = blockIdx.x / (QKVN / 64), nb = blockIdx.x % (QKVN / 64);
  const int m0 = mb * 64 + wv * 16, n0 = nb * 64;
  const unsigned short* arow = A + (size_t)(m0 + (lane & 15)) * HID;
  v8f acc[4] = {};
  if constexpr (PRE) {
    stage_b_async(&lb[0][0], Bh, QKVN, 0, n0, tid);
    wait_async0();
    __syncthreads();
    int cur = 0;
    for (int k0 = 0; k0 < HID; k0 += 32) {
      if (k0 + 32 < HID) stage_b_async(&lb[cur ^ 1][0], Bh, QKVN, k0 + 32, n0, tid);
      FragBF a;
      load_a_frag(a, arow, k0, lane);
#pragma unroll
      for (int j = 0; j < 4; ++j) {
        FragBF b;
        load_b_frag_lds(b, &lb[cur][0], j, lane);
        acc[j] = wmma_bf16(a, b, acc[j]);
      }
      wait_async0();
      __syncthreads();
      cur ^= 1;
    }
  } else {
    for (int k0 = 0; k0 < HID; k0 += 32) {
      FragBF a;
      load_a_frag(a, arow, k0, lane);
#pragma unroll
      for (int j = 0; j < 4; ++j) {
        FragBF b;
        load_b_frag_f32(b, Bf, QKVN, k0, n0 + 16 * j, lane);
        acc[j] = wmma_bf16(a, b, acc[j]);
      }
    }
  }
  const int half = lane >> 4, col = lane & 15;
#pragma unroll
  for (int j = 0; j < 4; ++j)
#pragma unroll
    for (int r = 0; r < 8; ++r)
      C[(size_t)(m0 + r + 8 * half) * QKVN + n0 + 16 * j + col] = acc[j][r];
}

// out = x + attn(bf16)[2048x768] @ w_out[768x768]
template <bool PRE>
__global__ void __launch_bounds__(128) k_gemm_outproj(const unsigned short* __restrict__ A,
                                                      const float* __restrict__ Bf,
                                                      const unsigned short* __restrict__ Bh,
                                                      const float* __restrict__ X,
                                                      float* __restrict__ O) {
  __shared__ alignas(16) unsigned short lb[2][32 * 64];
  const int tid = threadIdx.x;
  const int lane = tid & 31;
  const int wv = tid >> 5;
  const int mb = blockIdx.x / (HID / 64), nb = blockIdx.x % (HID / 64);
  const int m0 = mb * 64 + wv * 16, n0 = nb * 64;
  const unsigned short* arow = A + (size_t)(m0 + (lane & 15)) * HID;
  v8f acc[4] = {};
  if constexpr (PRE) {
    stage_b_async(&lb[0][0], Bh, HID, 0, n0, tid);
    wait_async0();
    __syncthreads();
    int cur = 0;
    for (int k0 = 0; k0 < HID; k0 += 32) {
      if (k0 + 32 < HID) stage_b_async(&lb[cur ^ 1][0], Bh, HID, k0 + 32, n0, tid);
      FragBF a;
      load_a_frag(a, arow, k0, lane);
#pragma unroll
      for (int j = 0; j < 4; ++j) {
        FragBF b;
        load_b_frag_lds(b, &lb[cur][0], j, lane);
        acc[j] = wmma_bf16(a, b, acc[j]);
      }
      wait_async0();
      __syncthreads();
      cur ^= 1;
    }
  } else {
    for (int k0 = 0; k0 < HID; k0 += 32) {
      FragBF a;
      load_a_frag(a, arow, k0, lane);
#pragma unroll
      for (int j = 0; j < 4; ++j) {
        FragBF b;
        load_b_frag_f32(b, Bf, HID, k0, n0 + 16 * j, lane);
        acc[j] = wmma_bf16(a, b, acc[j]);
      }
    }
  }
  const int half = lane >> 4, col = lane & 15;
#pragma unroll
  for (int j = 0; j < 4; ++j)
#pragma unroll
    for (int r = 0; r < 8; ++r) {
      const size_t idx = (size_t)(m0 + r + 8 * half) * HID + n0 + 16 * j + col;
      O[idx] = X[idx] + acc[j][r];
    }
}

// ---------------- flash attention ----------------
__global__ void __launch_bounds__(128) k_attention(const unsigned short* __restrict__ qb,
                                                   const unsigned short* __restrict__ kTb,
                                                   const unsigned short* __restrict__ vbuf,
                                                   unsigned short* __restrict__ attn) {
  __shared__ alignas(16) unsigned short plds[4][16 * 32];
  const int lane = threadIdx.x & 31;
  const int wv = threadIdx.x >> 5;
  const int gw = blockIdx.x * 4 + wv;              // 0..1535
  const int head = gw / (SEQ / 16);
  const int q0 = (gw % (SEQ / 16)) * 16;
  const int kv = head >> 2;                        // REP = 4
  const int half = lane >> 4;
  const int col = lane & 15;

  const unsigned short* Q = qb + ((size_t)head * SEQ + q0) * HD;
  const unsigned short* KT = kTb + (size_t)kv * HD * SEQ;
  const unsigned short* V = vbuf + (size_t)kv * SEQ * HD;

  FragBF qa0, qa1;
  {
    const unsigned short* qr = Q + (size_t)(lane & 15) * HD;
    load_a_frag(qa0, qr, 0, lane);
    load_a_frag(qa1, qr, 32, lane);
  }

  v8f O[4] = {};
  float m[8], l[8];
#pragma unroll
  for (int r = 0; r < 8; ++r) { m[r] = -1e30f; l[r] = 0.f; }

  const int kend = q0 + 16;
  for (int kb = 0; kb < kend; kb += 32) {
    const bool t1 = (kb + 16) < kend;
    v8f s0 = {}, s1 = {};
    {
      FragBF b;
      load_b_frag_bf16(b, KT, SEQ, 0, kb, lane);
      s0 = wmma_bf16(qa0, b, s0);
      load_b_frag_bf16(b, KT, SEQ, 32, kb, lane);
      s0 = wmma_bf16(qa1, b, s0);
      if (t1) {
        load_b_frag_bf16(b, KT, SEQ, 0, kb + 16, lane);
        s1 = wmma_bf16(qa0, b, s1);
        load_b_frag_bf16(b, KT, SEQ, 32, kb + 16, lane);
        s1 = wmma_bf16(qa1, b, s1);
      }
    }
#pragma unroll
    for (int r = 0; r < 8; ++r) {
      const int qrow = q0 + r + 8 * half;
      float a0 = s0[r] * 0.125f;
      if (kb + col > qrow) a0 = -1e30f;
      float a1 = t1 ? s1[r] * 0.125f : -1e30f;
      if (kb + 16 + col > qrow) a1 = -1e30f;
      float rm = fmaxf(a0, a1);
      rm = fmaxf(rm, __shfl_xor(rm, 1, 32));
      rm = fmaxf(rm, __shfl_xor(rm, 2, 32));
      rm = fmaxf(rm, __shfl_xor(rm, 4, 32));
      rm = fmaxf(rm, __shfl_xor(rm, 8, 32));
      const float nm = fmaxf(m[r], rm);
      const float sc = __expf(m[r] - nm);
      const float p0 = __expf(a0 - nm);
      const float p1 = __expf(a1 - nm);
      float rs = p0 + p1;
      rs += __shfl_xor(rs, 1, 32);
      rs += __shfl_xor(rs, 2, 32);
      rs += __shfl_xor(rs, 4, 32);
      rs += __shfl_xor(rs, 8, 32);
      l[r] = l[r] * sc + rs;
      m[r] = nm;
#pragma unroll
      for (int c = 0; c < 4; ++c) O[c][r] *= sc;
      const int pr = (r + 8 * half) * 32;
      plds[wv][pr + col] = f2bf(p0);
      plds[wv][pr + 16 + col] = f2bf(p1);
    }
    FragBF pa;
    load_a_frag(pa, &plds[wv][(lane & 15) * 32], 0, lane);
    const unsigned short* vrow = V + (size_t)(kb + 8 * half) * HD + col;
#pragma unroll
    for (int c = 0; c < 4; ++c) {
      FragBF vf;
#pragma unroll
      for (int s = 0; s < 8; ++s) {
        vf.u[s] = vrow[(size_t)s * HD + 16 * c];
        vf.u[8 + s] = vrow[(size_t)(16 + s) * HD + 16 * c];
      }
      O[c] = wmma_bf16(pa, vf, O[c]);
    }
  }
#pragma unroll
  for (int r = 0; r < 8; ++r) {
    const float inv = 1.f / l[r];
    unsigned short* orow = attn + (size_t)(q0 + r + 8 * half) * HID + head * HD;
#pragma unroll
    for (int c = 0; c < 4; ++c) orow[16 * c + col] = f2bf(O[c][r] * inv);
  }
}

// ---------------- router & scheduling ----------------
__global__ void k_router(const unsigned short* __restrict__ xn2, const float* __restrict__ wr,
                         int* __restrict__ t2i, float* __restrict__ t2w) {
  const int t = blockIdx.x * blockDim.x + threadIdx.x;
  if (t >= SEQ) return;
  float lg[NEXP];
#pragma unroll
  for (int e = 0; e < NEXP; ++e) lg[e] = 0.f;
  const unsigned short* xr = xn2 + (size_t)t * HID;
  for (int i = 0; i < HID; ++i) {
    const float xv = bf2f(xr[i]);
    const float* wrow = wr + i * NEXP;
#pragma unroll
    for (int e = 0; e < NEXP; ++e) lg[e] += xv * wrow[e];
  }
  float mx = lg[0];
#pragma unroll
  for (int e = 1; e < NEXP; ++e) mx = fmaxf(mx, lg[e]);
  float p[NEXP];
#pragma unroll
  for (int e = 0; e < NEXP; ++e) p[e] = __expf(lg[e] - mx);
  int e0 = 0; float v0 = p[0];
#pragma unroll
  for (int e = 1; e < NEXP; ++e) if (p[e] > v0) { v0 = p[e]; e0 = e; }
  int e1 = (e0 == 0) ? 1 : 0; float v1 = p[e1];
#pragma unroll
  for (int e = 0; e < NEXP; ++e) if (e != e0 && p[e] > v1) { v1 = p[e]; e1 = e; }
  const float s = v0 + v1;
  t2i[2 * t] = e0; t2i[2 * t + 1] = e1;
  t2w[2 * t] = v0 / s; t2w[2 * t + 1] = v1 / s;
}

__global__ void k_sched_init(int* __restrict__ sched, int* __restrict__ ltk, float* __restrict__ lw) {
  for (int i = threadIdx.x; i < SCH_INTS; i += 256) sched[i] = 0;
  for (int i = threadIdx.x; i < PADSLOTS; i += 256) { ltk[i] = 0; lw[i] = 0.f; }
}

__global__ void k_count(const int* __restrict__ t2i, int* __restrict__ sched) {
  const int g = blockIdx.x * blockDim.x + threadIdx.x;
  if (g < 2 * SEQ) atomicAdd(&sched[SCH_COUNT + t2i[g]], 1);
}

__global__ void k_scan(int* __restrict__ sched) {
  int off = 0, tb = 0;
  for (int e = 0; e < NEXP; ++e) {
    sched[SCH_PADOFF + e] = off;
    const int c = sched[SCH_COUNT + e];
    const int tt = (c + 15) >> 4;
    for (int i = 0; i < tt; ++i) {
      sched[SCH_TEXP + tb + i] = e;
      sched[SCH_TSLOT + tb + i] = off + i * 16;
    }
    tb += tt;
    off += tt * 16;
  }
  sched[SCH_TOTAL] = tb;
}

__global__ void k_fill(const int* __restrict__ t2i, const float* __restrict__ t2w,
                       int* __restrict__ sched, int* __restrict__ ltk, float* __restrict__ lw) {
  const int g = blockIdx.x * blockDim.x + threadIdx.x;
  if (g >= 2 * SEQ) return;
  const int e = t2i[g];
  const int pos = atomicAdd(&sched[SCH_CURSOR + e], 1);
  const int d = sched[SCH_PADOFF + e] + pos;
  ltk[d] = g >> 1;
  lw[d] = t2w[g];
}

// ---------------- grouped MoE GEMMs: one wave = 16(M) x 64(N) ----------------

template <bool PRE>
__global__ void __launch_bounds__(128) k_gateup(const unsigned short* __restrict__ xn2,
                                                const float* __restrict__ Bf,
                                                const unsigned short* __restrict__ Bh,
                                                const int* __restrict__ sched,
                                                const int* __restrict__ ltk,
                                                unsigned short* __restrict__ hbuf) {
  const int lane = threadIdx.x & 31;
  const int gw = blockIdx.x * 4 + (threadIdx.x >> 5);
  const int tile = gw / (FF / 64), cb = gw % (FF / 64);
  if (tile >= sched[SCH_TOTAL]) return;
  const int e = sched[SCH_TEXP + tile];
  const int slot0 = sched[SCH_TSLOT + tile];
  const int tok = ltk[slot0 + (lane & 15)];
  const unsigned short* arow = xn2 + (size_t)tok * HID;
  const float* Bfe = Bf + (size_t)e * HID * (2 * FF);
  const unsigned short* Bhe = Bh + (size_t)e * HID * (2 * FF);
  const int n0 = cb * 64;
  v8f g[4] = {}, u[4] = {};
  for (int k0 = 0; k0 < HID; k0 += 32) {
    FragBF a;
    load_a_frag(a, arow, k0, lane);
#pragma unroll
    for (int j = 0; j < 4; ++j) {
      FragBF b;
      load_b<PRE>(b, Bfe, Bhe, 2 * FF, k0, n0 + 16 * j, lane);
      g[j] = wmma_bf16(a, b, g[j]);
      load_b<PRE>(b, Bfe, Bhe, 2 * FF, k0, FF + n0 + 16 * j, lane);
      u[j] = wmma_bf16(a, b, u[j]);
    }
  }
  const int half = lane >> 4, col = lane & 15;
#pragma unroll
  for (int j = 0; j < 4; ++j)
#pragma unroll
    for (int r = 0; r < 8; ++r) {
      const float gv = g[j][r];
      const float hv = (gv / (1.f + __expf(-gv))) * u[j][r];
      hbuf[(size_t)(slot0 + r + 8 * half) * FF + n0 + 16 * j + col] = f2bf(hv);
    }
}

template <bool PRE>
__global__ void __launch_bounds__(128) k_down(const unsigned short* __restrict__ hbuf,
                                              const float* __restrict__ Bf,
                                              const unsigned short* __restrict__ Bh,
                                              const int* __restrict__ sched,
                                              const int* __restrict__ ltk,
                                              const float* __restrict__ lw,
                                              float* __restrict__ O) {
  const int lane = threadIdx.x & 31;
  const int gw = blockIdx.x * 4 + (threadIdx.x >> 5);
  const int tile = gw / (HID / 64), cb = gw % (HID / 64);
  if (tile >= sched[SCH_TOTAL]) return;
  const int e = sched[SCH_TEXP + tile];
  const int slot0 = sched[SCH_TSLOT + tile];
  const unsigned short* arow = hbuf + (size_t)(slot0 + (lane & 15)) * FF;
  const float* Bfe = Bf + (size_t)e * FF * HID;
  const unsigned short* Bhe = Bh + (size_t)e * FF * HID;
  const int n0 = cb * 64;
  v8f acc[4] = {};
  for (int k0 = 0; k0 < FF; k0 += 32) {
    FragBF a;
    load_a_frag(a, arow, k0, lane);
#pragma unroll
    for (int j = 0; j < 4; ++j) {
      FragBF b;
      load_b<PRE>(b, Bfe, Bhe, HID, k0, n0 + 16 * j, lane);
      acc[j] = wmma_bf16(a, b, acc[j]);
    }
    if (PRE) __builtin_prefetch(Bhe + (size_t)(k0 + 32 + lane) * HID + n0, 0, 1);
  }
  const int half = lane >> 4, col = lane & 15;
#pragma unroll
  for (int j = 0; j < 4; ++j)
#pragma unroll
    for (int r = 0; r < 8; ++r) {
      const int slot = slot0 + r + 8 * half;
      atomicAdd(&O[(size_t)ltk[slot] * HID + n0 + 16 * j + col], lw[slot] * acc[j][r]);
    }
}

// ---------------- workspace layout (bytes) ----------------
constexpr size_t alignup(size_t x) { return (x + 255) & ~(size_t)255; }
constexpr size_t OFF_XN1 = 0;
constexpr size_t OFF_QKV = alignup(OFF_XN1 + (size_t)SEQ * HID * 2);
constexpr size_t OFF_QB  = alignup(OFF_QKV + (size_t)SEQ * QKVN * 4);
constexpr size_t OFF_KT  = alignup(OFF_QB + (size_t)NQ * SEQ * HD * 2);
constexpr size_t OFF_VB  = alignup(OFF_KT + (size_t)NKV * HD * SEQ * 2);
constexpr size_t OFF_ATT = alignup(OFF_VB + (size_t)NKV * SEQ * HD * 2);
constexpr size_t OFF_XN2 = alignup(OFF_ATT + (size_t)SEQ * HID * 2);
constexpr size_t OFF_T2I = alignup(OFF_XN2 + (size_t)SEQ * HID * 2);
constexpr size_t OFF_T2W = alignup(OFF_T2I + (size_t)2 * SEQ * 4);
constexpr size_t OFF_SCH = alignup(OFF_T2W + (size_t)2 * SEQ * 4);
constexpr size_t OFF_LTK = alignup(OFF_SCH + (size_t)SCH_INTS * 4);
constexpr size_t OFF_LW  = alignup(OFF_LTK + (size_t)PADSLOTS * 4);
constexpr size_t OFF_HB  = alignup(OFF_LW + (size_t)PADSLOTS * 4);
constexpr size_t WS_BASE = alignup(OFF_HB + (size_t)PADSLOTS * FF * 2);  // ~37 MB
// optional pre-converted bf16 weights (used only if ws_size is big enough)
constexpr size_t OFF_WQKVH = WS_BASE;
constexpr size_t OFF_WOUTH = alignup(OFF_WQKVH + (size_t)HID * QKVN * 2);
constexpr size_t OFF_WGUH  = alignup(OFF_WOUTH + (size_t)HID * HID * 2);
constexpr size_t OFF_WDNH  = alignup(OFF_WGUH + (size_t)NEXP * HID * 2 * FF * 2);
constexpr size_t WS_PRE    = alignup(OFF_WDNH + (size_t)NEXP * FF * HID * 2);  // ~153 MB

extern "C" void kernel_launch(void* const* d_in, const int* in_sizes, int n_in,
                              void* d_out, int out_size, void* d_ws, size_t ws_size,
                              hipStream_t stream) {
  (void)in_sizes; (void)n_in; (void)out_size;
  const float* x        = (const float*)d_in[0];
  const float* norm1_w  = (const float*)d_in[1];
  const float* w_qkv    = (const float*)d_in[2];
  const float* w_out    = (const float*)d_in[3];
  const float* norm2_w  = (const float*)d_in[4];
  const float* w_router = (const float*)d_in[5];
  const float* w_gu     = (const float*)d_in[6];
  const float* w_dn     = (const float*)d_in[7];
  float* out = (float*)d_out;

  char* ws = (char*)d_ws;
  unsigned short* xn1  = (unsigned short*)(ws + OFF_XN1);
  float*          qkv  = (float*)(ws + OFF_QKV);
  unsigned short* qb   = (unsigned short*)(ws + OFF_QB);
  unsigned short* kT   = (unsigned short*)(ws + OFF_KT);
  unsigned short* vb   = (unsigned short*)(ws + OFF_VB);
  unsigned short* attn = (unsigned short*)(ws + OFF_ATT);
  unsigned short* xn2  = (unsigned short*)(ws + OFF_XN2);
  int*            t2i  = (int*)(ws + OFF_T2I);
  float*          t2w  = (float*)(ws + OFF_T2W);
  int*            sch  = (int*)(ws + OFF_SCH);
  int*            ltk  = (int*)(ws + OFF_LTK);
  float*          lw   = (float*)(ws + OFF_LW);
  unsigned short* hbuf = (unsigned short*)(ws + OFF_HB);
  unsigned short* wqkvh = (unsigned short*)(ws + OFF_WQKVH);
  unsigned short* wouth = (unsigned short*)(ws + OFF_WOUTH);
  unsigned short* wguh  = (unsigned short*)(ws + OFF_WGUH);
  unsigned short* wdnh  = (unsigned short*)(ws + OFF_WDNH);

  const bool pre = (ws_size >= WS_PRE);  // ws_size constant per run -> deterministic

  if (pre) {
    // one-shot f32 -> bf16 weight conversion (bandwidth-trivial, saves inner-loop VALU)
    k_cvt_bf16<<<1024, 256, 0, stream>>>(w_qkv, wqkvh, HID * QKVN / 4);
    k_cvt_bf16<<<1024, 256, 0, stream>>>(w_out, wouth, HID * HID / 4);
    k_cvt_bf16<<<2048, 256, 0, stream>>>(w_gu, wguh, NEXP * HID * 2 * FF / 4);
    k_cvt_bf16<<<2048, 256, 0, stream>>>(w_dn, wdnh, NEXP * FF * HID / 4);
  }

  // 1. pre-norm 1
  k_rmsnorm<<<SEQ, 256, 0, stream>>>(x, norm1_w, xn1);
  // 2. QKV projection (WMMA + async LDS staging): (2048/64)*(1152/64) = 576 blocks
  if (pre) k_gemm_qkv<true><<<(SEQ / 64) * (QKVN / 64), 128, 0, stream>>>(xn1, nullptr, wqkvh, qkv);
  else     k_gemm_qkv<false><<<(SEQ / 64) * (QKVN / 64), 128, 0, stream>>>(xn1, w_qkv, nullptr, qkv);
  // 3. RoPE + bf16 packing (q row-major, k transposed, v row-major)
  k_rope<<<(SEQ * 18 + 127) / 128, 128, 0, stream>>>(qkv, qb, kT, vb);
  // 4. flash attention (WMMA): 12 heads * 128 q-tiles = 1536 waves
  k_attention<<<NQ * (SEQ / 16) / 4, 128, 0, stream>>>(qb, kT, vb, attn);
  // 5. output projection + residual (WMMA + async LDS staging): writes h into d_out
  if (pre) k_gemm_outproj<true><<<(SEQ / 64) * (HID / 64), 128, 0, stream>>>(attn, nullptr, wouth, x, out);
  else     k_gemm_outproj<false><<<(SEQ / 64) * (HID / 64), 128, 0, stream>>>(attn, w_out, nullptr, x, out);
  // 6. pre-norm 2 (reads h from d_out)
  k_rmsnorm<<<SEQ, 256, 0, stream>>>(out, norm2_w, xn2);
  // 7. router top-2
  k_router<<<SEQ / 256, 256, 0, stream>>>(xn2, w_router, t2i, t2w);
  // 8-11. build padded per-expert token lists
  k_sched_init<<<1, 256, 0, stream>>>(sch, ltk, lw);
  k_count<<<(2 * SEQ) / 256, 256, 0, stream>>>(t2i, sch);
  k_scan<<<1, 1, 0, stream>>>(sch);
  k_fill<<<(2 * SEQ) / 256, 256, 0, stream>>>(t2i, t2w, sch, ltk, lw);
  // 12. grouped gate/up GEMM + SiLU (WMMA), worst-case grid, early-exit by total tiles
  if (pre) k_gateup<true><<<MAXTILES * (FF / 64) / 4, 128, 0, stream>>>(xn2, nullptr, wguh, sch, ltk, hbuf);
  else     k_gateup<false><<<MAXTILES * (FF / 64) / 4, 128, 0, stream>>>(xn2, w_gu, nullptr, sch, ltk, hbuf);
  // 13. grouped down GEMM + weighted scatter-add into d_out (WMMA)
  if (pre) k_down<true><<<MAXTILES * (HID / 64) / 4, 128, 0, stream>>>(hbuf, nullptr, wdnh, sch, ltk, lw, out);
  else     k_down<false><<<MAXTILES * (HID / 64) / 4, 128, 0, stream>>>(hbuf, w_dn, nullptr, sch, ltk, lw, out);
}
